// ExplainableHeterogeneousGAT_90202903151259
// MI455X (gfx1250) — matrix operand order repaired and checked by reading, not compiled, not run on hardware
//
#include <hip/hip_runtime.h>
#include <hip/hip_bf16.h>
#include <math.h>

// ---------------- problem constants (fixed by reference) ----------------
#define NG   30000
#define ND   8000
#define Dh   128     // hidden dim
#define NH   4       // heads
#define HC   512     // NH * C
#define E_GG 200000
#define E_GD 100000
#define E_DG 100000

typedef __attribute__((ext_vector_type(16))) __bf16 v16bf;
typedef __attribute__((ext_vector_type(8)))  float  v8f;

// ---------------- device helpers ----------------
static __device__ __forceinline__ unsigned short f2bf(float f) {
  unsigned int u = __float_as_uint(f);
  unsigned int r = (u + 0x7FFFu + ((u >> 16) & 1u)) >> 16;  // RNE
  return (unsigned short)r;
}
static __device__ __forceinline__ float elu1(float x)  { return x > 0.f ? x : expf(x) - 1.f; }
static __device__ __forceinline__ float lrelu(float x) { return x > 0.f ? x : 0.2f * x; }

static __device__ __forceinline__ void atomicMaxF(float* addr, float v) {
  if (v >= 0.f) atomicMax((int*)addr, __float_as_int(v));
  else          atomicMin((unsigned int*)addr, __float_as_uint(v));
}

// ---------------- elementwise kernels ----------------
__global__ void k_cvt_bf16(const float* __restrict__ src, unsigned short* __restrict__ dst, int n) {
  for (int i = blockIdx.x * blockDim.x + threadIdx.x; i < n; i += gridDim.x * blockDim.x)
    dst[i] = f2bf(src[i]);
}

// Pack W [K,N] f32 row-major into WMMA-B fragment-friendly tiles:
//   Wp[((k/32)*N + n)*32 + (k%32)] = bf16(W[k,n])
// so a lane's 16 needed K-values for a column are contiguous (two b128 loads).
__global__ void k_pack_w(const float* __restrict__ W, unsigned short* __restrict__ Wp, int K, int N) {
  int t = blockIdx.x * blockDim.x + threadIdx.x;
  if (t >= K * N) return;
  int k = t / N, n = t - k * N;
  Wp[((size_t)(k >> 5) * N + n) * 32 + (k & 31)] = f2bf(W[t]);
}

__global__ void k_fill(float* p, float val, int n) {
  for (int i = blockIdx.x * blockDim.x + threadIdx.x; i < n; i += gridDim.x * blockDim.x) p[i] = val;
}

// ---------------- WMMA GEMM: C[M,N] = act(A_bf16[M,K] @ W + bias) ----------------
// block = 256 threads = 8 wave32; wave w computes a 16 x (16*TN) tile:
//   rows  [blockIdx.y*16, +16), cols [blockIdx.x*128*TN + w*16*TN, +16*TN)
// A fragment is loaded once per k-step and reused across TN B tiles.
// Tail rows: A row index is clamped (a clamped row only feeds output rows >= M,
// which are never stored), so the k-loop has no EXEC manipulation at all.
union FragAB { uint4 q[2]; v16bf v; };

template <int TN>
__global__ void k_wmma_gemm(const unsigned short* __restrict__ A,
                            const unsigned short* __restrict__ Wp,
                            const float* __restrict__ bias,
                            float* __restrict__ Cmat,
                            int M, int K, int N, int act) {
  const int lane = threadIdx.x & 31;
  const int wave = threadIdx.x >> 5;
  const int n0   = blockIdx.x * (128 * TN) + wave * (16 * TN);
  const int m0   = blockIdx.y * 16;
  const int lm   = lane & 15;
  const int hi   = lane >> 4;            // half-wave select per ISA fragment layout
  int m = m0 + lm;                       // A row for this lane
  if (m >= M) m = M - 1;                 // clamp (see note above)
  const int kbA = hi ? 16 : 0;           // A: lanes 16-31 hold K+16 block
  const int kbB = hi ? 16 : 0;           // B: lanes 16-31 hold K+16 block

  v8f acc[TN];
#pragma unroll
  for (int t = 0; t < TN; ++t) acc[t] = (v8f){0.f, 0.f, 0.f, 0.f, 0.f, 0.f, 0.f, 0.f};

  const unsigned short* arow = A + (size_t)m * K;
  for (int kk = 0; kk < K; kk += 32) {
    FragAB a;
    const uint4* ap = (const uint4*)(arow + kk + kbA);
    a.q[0] = ap[0];                      // K = kk+kbA   .. +7
  a.q[1] = ap[1];                      // K = kk+8+kbA .. +7
    __builtin_prefetch(arow + kk + 32, 0, 1);
    const unsigned short* btile = Wp + (size_t)(kk >> 5) * N * 32 + kbB;
#pragma unroll
    for (int t = 0; t < TN; ++t) {
      const uint4* bp = (const uint4*)(btile + (size_t)(n0 + t * 16 + lm) * 32);
      FragAB b;
      b.q[0] = bp[0];
      b.q[1] = bp[1];
      acc[t] = __builtin_amdgcn_wmma_f32_16x16x32_bf16(false, a.v, false, b.v,
                                                       (short)0, acc[t], false, false);
    }
  }

  const int rbase = m0 + (hi ? 8 : 0);   // D layout: VGPR r -> row r (+8 for upper lanes)
#pragma unroll
  for (int t = 0; t < TN; ++t) {
    const int n  = n0 + t * 16 + lm;
    const float bv = bias ? bias[n] : 0.f;
#pragma unroll
    for (int r = 0; r < 8; ++r) {
      int row = rbase + r;
      if (row < M) {
        float v = acc[t][r] + bv;
        if (act) v = elu1(v);
        Cmat[(size_t)row * N + n] = v;
      }
    }
  }
}

// ---------------- attention helpers ----------------
// v[h,i] = sum_c W[i, h*128+c] * a[h,c]   (folds attention vector through W)
__global__ void k_fold(const float* __restrict__ W, const float* __restrict__ avec,
                       float* __restrict__ out) {
  int t = blockIdx.x * blockDim.x + threadIdx.x;
  if (t >= NH * Dh) return;
  int h = t >> 7, i = t & 127;
  const float* wrow = W + (size_t)i * HC + h * Dh;
  const float* ah   = avec + h * Dh;
  float s = 0.f;
  for (int c = 0; c < Dh; ++c) s += wrow[c] * ah[c];
  out[t] = s;
}

// out[n,h] = x[n,:] . v[h,:]
__global__ void k_att_proj(const float* __restrict__ x, const float* __restrict__ v,
                           float* __restrict__ out, int N) {
  int t = blockIdx.x * blockDim.x + threadIdx.x;
  if (t >= N * NH) return;
  int n = t >> 2, h = t & 3;
  const float* xr = x + (size_t)n * Dh;
  const float* vh = v + h * Dh;
  float s = 0.f;
  for (int i = 0; i < Dh; ++i) s += xr[i] * vh[i];
  out[t] = s;
}

__global__ void k_edge_max(const float* __restrict__ as_, const float* __restrict__ ad_,
                           const int* __restrict__ src, const int* __restrict__ dst,
                           float* __restrict__ m, int E) {
  int t = blockIdx.x * blockDim.x + threadIdx.x;
  if (t >= E * NH) return;
  int e = t >> 2, h = t & 3;
  float v = lrelu(as_[src[e] * NH + h] + ad_[dst[e] * NH + h]);
  atomicMaxF(&m[dst[e] * NH + h], v);
}

__global__ void k_clamp_m(float* m, int n) {
  int t = blockIdx.x * blockDim.x + threadIdx.x;
  if (t >= n) return;
  float v = m[t];
  if (!(v > -1e30f)) v = 0.f;   // empty segment: -inf -> 0 (matches isfinite clamp)
  m[t] = v;
}

__global__ void k_edge_den(const float* __restrict__ as_, const float* __restrict__ ad_,
                           const int* __restrict__ src, const int* __restrict__ dst,
                           const float* __restrict__ m, float* __restrict__ den,
                           float* __restrict__ exbuf, int E) {
  int t = blockIdx.x * blockDim.x + threadIdx.x;
  if (t >= E * NH) return;
  int e = t >> 2, h = t & 3;
  int d = dst[e];
  float v  = lrelu(as_[src[e] * NH + h] + ad_[d * NH + h]);
  float ex = expf(v - m[d * NH + h]);
  exbuf[t] = ex;
  unsafeAtomicAdd(&den[d * NH + h], ex);      // global_atomic_add_f32
}

// one wave32 per edge: lane l owns columns [4l,4l+4) for each head
__global__ void k_edge_scatter(const float* __restrict__ hs, const int* __restrict__ src,
                               const int* __restrict__ dst, const float* __restrict__ exbuf,
                               const float* __restrict__ den, float* __restrict__ accum, int E) {
  int gid = blockIdx.x * blockDim.x + threadIdx.x;
  int e = gid >> 5, l = gid & 31;
  if (e >= E) return;
  int s = src[e], d = dst[e];
  const float* hrow = hs + (size_t)s * HC;
  float*       arow = accum + (size_t)d * HC;
#pragma unroll
  for (int h = 0; h < NH; ++h) {
    float alpha = exbuf[e * NH + h] / (den[d * NH + h] + 1e-16f);
    const float4 hv = *(const float4*)(hrow + h * Dh + l * 4);
    float* ap = arow + h * Dh + l * 4;
    unsafeAtomicAdd(ap + 0, hv.x * alpha);
    unsafeAtomicAdd(ap + 1, hv.y * alpha);
    unsafeAtomicAdd(ap + 2, hv.z * alpha);
    unsafeAtomicAdd(ap + 3, hv.w * alpha);
  }
}

// out[n,c] = scale * (mean_h accum[n,h,c] + b1[c] (+ b2[c]))
__global__ void k_finalize(const float* __restrict__ accum, const float* __restrict__ b1,
                           const float* __restrict__ b2, float scale,
                           float* __restrict__ out, int N) {
  int t = blockIdx.x * blockDim.x + threadIdx.x;
  if (t >= N * Dh) return;
  int n = t >> 7, c = t & 127;
  const float* ar = accum + (size_t)n * HC;
  float s  = 0.25f * (ar[c] + ar[Dh + c] + ar[2 * Dh + c] + ar[3 * Dh + c]);
  float bb = b1[c] + (b2 ? b2[c] : 0.f);
  out[t] = scale * (s + bb);
}

// one block per channel
__global__ void k_bn_stats(const float* __restrict__ x, float* __restrict__ mv, int N) {
  int c = blockIdx.x;
  float s = 0.f, s2 = 0.f;
  for (int n = threadIdx.x; n < N; n += blockDim.x) {
    float v = x[(size_t)n * Dh + c]; s += v; s2 += v * v;
  }
  __shared__ float sh[256], sh2[256];
  sh[threadIdx.x] = s; sh2[threadIdx.x] = s2;
  __syncthreads();
  for (int st = blockDim.x >> 1; st > 0; st >>= 1) {
    if (threadIdx.x < (unsigned)st) { sh[threadIdx.x] += sh[threadIdx.x + st]; sh2[threadIdx.x] += sh2[threadIdx.x + st]; }
    __syncthreads();
  }
  if (threadIdx.x == 0) {
    float mu = sh[0] / N;
    mv[c]      = mu;
    mv[Dh + c] = sh2[0] / N - mu * mu;
  }
}

__global__ void k_bn_apply(const float* __restrict__ tmp, const float* __restrict__ g,
                           const float* __restrict__ b, const float* __restrict__ mv,
                           float* __restrict__ x, int residual, int N) {
  int t = blockIdx.x * blockDim.x + threadIdx.x;
  if (t >= N * Dh) return;
  int c = t & 127;
  float mu = mv[c], var = mv[Dh + c];
  float y = (tmp[t] - mu) * rsqrtf(var + 1e-5f) * g[c] + b[c];
  y = elu1(y);
  x[t] = residual ? x[t] + y : y;
}

// ---------------- host orchestration ----------------
extern "C" void kernel_launch(void* const* d_in, const int* in_sizes, int n_in,
                              void* d_out, int out_size, void* d_ws, size_t ws_size,
                              hipStream_t stream) {
  (void)in_sizes; (void)n_in; (void)out_size; (void)ws_size;

  const float* x_gene = (const float*)d_in[0];
  const float* x_drug = (const float*)d_in[1];
  const int* src_gg = (const int*)d_in[2];
  const int* dst_gg = (const int*)d_in[3];
  const int* src_gd = (const int*)d_in[4];
  const int* dst_gd = (const int*)d_in[5];
  const int* src_dg = (const int*)d_in[6];
  const int* dst_dg = (const int*)d_in[7];
  // params flattened in setup_inputs() insertion order:
  // 8:in_gene.W 9:in_gene.b 10:in_drug.W 11:in_drug.b
  // 12+l*15: layer l -> gg{Wsrc,Wdst,asrc,adst,b} gd{...} dg{...}
  // 42+l*4: bn l -> gene.g gene.b drug.g drug.b
  // 50:out_gene.W 51:out_gene.b 52:out_drug.W 53:out_drug.b
  auto Pf = [&](int i) { return (const float*)d_in[i]; };

  // ---- workspace carve-up ----
  char* w = (char*)d_ws;
  size_t o = 0;
  auto alloc = [&](size_t bytes) -> void* {
    void* p = w + o; o = (o + bytes + 255) & ~(size_t)255; return p;
  };
  float*          xg      = (float*)alloc((size_t)NG * Dh * 4);
  float*          xd      = (float*)alloc((size_t)ND * Dh * 4);
  unsigned short* xg_bf   = (unsigned short*)alloc((size_t)NG * 256 * 2);
  unsigned short* xd_bf   = (unsigned short*)alloc((size_t)ND * Dh * 2);
  unsigned short* Wp      = (unsigned short*)alloc((size_t)256 * 512 * 2);
  float*          accum_g = (float*)alloc((size_t)NG * HC * 4);
  float*          accum_d = (float*)alloc((size_t)ND * HC * 4);
  float*          hsb     = (float*)alloc((size_t)NG * HC * 4);
  float*          tmp_g   = (float*)alloc((size_t)NG * Dh * 4);
  float*          tmp_d   = (float*)alloc((size_t)ND * Dh * 4);
  float*          a_s     = (float*)alloc((size_t)NG * NH * 4);
  float*          a_d     = (float*)alloc((size_t)NG * NH * 4);
  float*          mbuf    = (float*)alloc((size_t)NG * NH * 4);
  float*          den     = (float*)alloc((size_t)NG * NH * 4);
  float*          exbuf   = (float*)alloc((size_t)E_GG * NH * 4);
  float*          vfold   = (float*)alloc((size_t)2 * NH * Dh * 4);
  float*          mv      = (float*)alloc((size_t)2 * Dh * 4);

  auto gemm = [&](const unsigned short* A, const float* Wf, const float* bias, float* C,
                  int M, int K, int N, int act) {
    k_pack_w<<<(K * N + 255) / 256, 256, 0, stream>>>(Wf, Wp, K, N);
    if (N == 512) {
      dim3 grid(1, (M + 15) / 16);                    // TN=4: 8 waves * 64 cols = 512
      k_wmma_gemm<4><<<grid, 256, 0, stream>>>(A, Wp, bias, C, M, K, N, act);
    } else {
      dim3 grid(N / 128, (M + 15) / 16);              // TN=1: 8 waves * 16 cols = 128
      k_wmma_gemm<1><<<grid, 256, 0, stream>>>(A, Wp, bias, C, M, K, N, act);
    }
  };

  auto relation = [&](const float* xsrc, const unsigned short* xsrc_bf, int Ns,
                      const float* xdst, int Nd2,
                      const int* src, const int* dst, int E,
                      const float* Wsrc, const float* Wdst,
                      const float* asrc, const float* adst, float* accum) {
    gemm(xsrc_bf, Wsrc, nullptr, hsb, Ns, Dh, HC, 0);                       // hs = x_src @ Wsrc
    k_fold<<<2, 256, 0, stream>>>(Wsrc, asrc, vfold);                       // fold asrc through Wsrc
    k_fold<<<2, 256, 0, stream>>>(Wdst, adst, vfold + NH * Dh);             // fold adst through Wdst
    k_att_proj<<<(Ns * NH + 255) / 256, 256, 0, stream>>>(xsrc, vfold, a_s, Ns);
    k_att_proj<<<(Nd2 * NH + 255) / 256, 256, 0, stream>>>(xdst, vfold + NH * Dh, a_d, Nd2);
    k_fill<<<256, 256, 0, stream>>>(mbuf, -INFINITY, Nd2 * NH);
    hipMemsetAsync(den, 0, (size_t)Nd2 * NH * 4, stream);
    k_edge_max<<<(E * NH + 255) / 256, 256, 0, stream>>>(a_s, a_d, src, dst, mbuf, E);
    k_clamp_m<<<(Nd2 * NH + 255) / 256, 256, 0, stream>>>(mbuf, Nd2 * NH);
    k_edge_den<<<(E * NH + 255) / 256, 256, 0, stream>>>(a_s, a_d, src, dst, mbuf, den, exbuf, E);
    k_edge_scatter<<<(E * 32 + 255) / 256, 256, 0, stream>>>(hsb, src, dst, exbuf, den, accum, E);
  };

  // ---- input linears (ELU) ----
  k_cvt_bf16<<<2048, 256, 0, stream>>>(x_gene, xg_bf, NG * 256);
  gemm(xg_bf, Pf(8), Pf(9), xg, NG, 256, Dh, 1);
  k_cvt_bf16<<<2048, 256, 0, stream>>>(x_drug, xd_bf, ND * Dh);
  gemm(xd_bf, Pf(10), Pf(11), xd, ND, Dh, Dh, 1);

  // ---- 2 hetero GAT layers ----
  for (int l = 0; l < 2; ++l) {
    int off = 12 + l * 15;
    const float *gg_Wsrc = Pf(off + 0),  *gg_Wdst = Pf(off + 1),  *gg_as = Pf(off + 2),
                *gg_ad = Pf(off + 3),    *gg_b = Pf(off + 4);
    const float *gd_Wsrc = Pf(off + 5),  *gd_Wdst = Pf(off + 6),  *gd_as = Pf(off + 7),
                *gd_ad = Pf(off + 8),    *gd_b = Pf(off + 9);
    const float *dg_Wsrc = Pf(off + 10), *dg_Wdst = Pf(off + 11), *dg_as = Pf(off + 12),
                *dg_ad = Pf(off + 13),   *dg_b = Pf(off + 14);
    int bo = 42 + l * 4;
    const float *bng_g = Pf(bo + 0), *bng_b = Pf(bo + 1), *bnd_g = Pf(bo + 2), *bnd_b = Pf(bo + 3);

    hipMemsetAsync(accum_g, 0, (size_t)NG * HC * 4, stream);
    hipMemsetAsync(accum_d, 0, (size_t)ND * HC * 4, stream);
    k_cvt_bf16<<<2048, 256, 0, stream>>>(xg, xg_bf, NG * Dh);
    k_cvt_bf16<<<2048, 256, 0, stream>>>(xd, xd_bf, ND * Dh);

    relation(xg, xg_bf, NG, xg, NG, src_gg, dst_gg, E_GG, gg_Wsrc, gg_Wdst, gg_as, gg_ad, accum_g);
    relation(xd, xd_bf, ND, xg, NG, src_dg, dst_dg, E_DG, dg_Wsrc, dg_Wdst, dg_as, dg_ad, accum_g);
    relation(xg, xg_bf, NG, xd, ND, src_gd, dst_gd, E_GD, gd_Wsrc, gd_Wdst, gd_as, gd_ad, accum_d);

    // ng = 0.5*(o_gg + o_dg); nd = o_gd   (head-mean + biases)
    k_finalize<<<(NG * Dh + 255) / 256, 256, 0, stream>>>(accum_g, gg_b, dg_b, 0.5f, tmp_g, NG);
    k_finalize<<<(ND * Dh + 255) / 256, 256, 0, stream>>>(accum_d, gd_b, nullptr, 1.0f, tmp_d, ND);
    k_bn_stats<<<Dh, 256, 0, stream>>>(tmp_g, mv, NG);
    k_bn_apply<<<(NG * Dh + 255) / 256, 256, 0, stream>>>(tmp_g, bng_g, bng_b, mv, xg, l, NG);
    k_bn_stats<<<Dh, 256, 0, stream>>>(tmp_d, mv, ND);
    k_bn_apply<<<(ND * Dh + 255) / 256, 256, 0, stream>>>(tmp_d, bnd_g, bnd_b, mv, xd, l, ND);
  }

  // ---- output linears ----
  float* out_g = (float*)d_out;
  float* out_d = out_g + (size_t)NG * Dh;
  k_cvt_bf16<<<2048, 256, 0, stream>>>(xg, xg_bf, NG * Dh);
  gemm(xg_bf, Pf(50), Pf(51), out_g, NG, Dh, Dh, 0);
  k_cvt_bf16<<<2048, 256, 0, stream>>>(xd, xd_bf, ND * Dh);
  gemm(xd_bf, Pf(52), Pf(53), out_d, ND, Dh, Dh, 0);
}
